// net5_18098992185369
// MI455X (gfx1250) — matrix operand order repaired and checked by reading, not compiled, not run on hardware
//
#include <hip/hip_runtime.h>
#include <math.h>

// ---------------- problem constants ----------------
#define T_STEPS 512
#define BATCH   64
#define IN      256
#define HID     512
#define OUTF    64
#define G3      (3 * HID)            // 1536
#define M1      (T_STEPS * BATCH)    // 32768 rows of x / activity
#define NWG2    16                   // workgroups in recurrent kernel
#define OUT_ELEMS (T_STEPS * BATCH * OUTF)   // 2097152 floats (first output)

// ---------------- workspace layout (bytes) ----------------
#define H_OFF    256                 // h double buffer (bf16), 128 KB
#define XB_OFF   0x00040000u         // x bf16:      32768*256*2  = 16 MB
#define W2B_OFF  0x01040000u         // x2h_w bf16:  1536*256*2   = 768 KB
#define FCB_OFF  0x01100000u         // fc_w bf16:   64*512*2     = 64 KB
#define ACTB_OFF 0x01200000u         // act bf16:    32768*512*2  = 32 MB
#define XPB_OFF  0x03400000u         // x_proj bf16: 32768*1536*2 = 96 MB

typedef __attribute__((ext_vector_type(16))) __bf16 v16bf;
typedef __attribute__((ext_vector_type(8)))  __bf16 bf16x8;
typedef __attribute__((ext_vector_type(4)))  __bf16 bf16x4;
typedef __attribute__((ext_vector_type(8)))  float  v8f;

// ---------------- helpers ----------------
__device__ __forceinline__ void ld8(const __bf16* p, v16bf& f, int base) {
  bf16x8 v = *reinterpret_cast<const bf16x8*>(p);
#pragma unroll
  for (int i = 0; i < 8; ++i) f[base + i] = v[i];
}

__device__ __forceinline__ void ld16(const __bf16* p, v16bf& f) {
  ld8(p, f, 0);
  ld8(p + 8, f, 8);
}

__device__ __forceinline__ void cvt16(const float* p, v16bf& f) {
#pragma unroll
  for (int h = 0; h < 2; ++h) {
    const float4 a = *reinterpret_cast<const float4*>(p + h * 8);
    const float4 b = *reinterpret_cast<const float4*>(p + h * 8 + 4);
    f[h * 8 + 0] = (__bf16)a.x; f[h * 8 + 1] = (__bf16)a.y;
    f[h * 8 + 2] = (__bf16)a.z; f[h * 8 + 3] = (__bf16)a.w;
    f[h * 8 + 4] = (__bf16)b.x; f[h * 8 + 5] = (__bf16)b.y;
    f[h * 8 + 6] = (__bf16)b.z; f[h * 8 + 7] = (__bf16)b.w;
  }
}

__device__ __forceinline__ void cvtstore4(const float* src, __bf16* dst, int i) {
  const float4 v = reinterpret_cast<const float4*>(src)[i];
  bf16x4 o;
  o[0] = (__bf16)v.x; o[1] = (__bf16)v.y;
  o[2] = (__bf16)v.z; o[3] = (__bf16)v.w;
  reinterpret_cast<bf16x4*>(dst)[i] = o;
}

__device__ __forceinline__ v8f wmma_bf16(v16bf a, v16bf b, v8f c) {
  return __builtin_amdgcn_wmma_f32_16x16x32_bf16(
      false, a, false, b, (short)0, c, false, false);
}

__device__ __forceinline__ float sigmoidf_(float x) {
  return 1.0f / (1.0f + __expf(-x));
}

__device__ __forceinline__ float tanhf_(float x) {
  // tanh(x) = 1 - 2/(exp(2x)+1): one v_exp_f32 + few FMAs (short serial chain)
  return 1.0f - 2.0f / (__expf(2.0f * x) + 1.0f);
}

// ---------------- kernel 0: one-shot bf16 conversion + init ----------------
__global__ __launch_bounds__(256) void k_prep(
    const float* __restrict__ x, const float* __restrict__ w2,
    const float* __restrict__ fw, unsigned short* __restrict__ xb_u16,
    unsigned short* __restrict__ w2b_u16, unsigned short* __restrict__ fcb_u16,
    unsigned* __restrict__ cnt, unsigned* __restrict__ hz) {
  __bf16* xb  = reinterpret_cast<__bf16*>(xb_u16);
  __bf16* w2b = reinterpret_cast<__bf16*>(w2b_u16);
  __bf16* fcb = reinterpret_cast<__bf16*>(fcb_u16);
  if (blockIdx.x == 0) {
    if (threadIdx.x == 0) *cnt = 0u;
    for (int i = threadIdx.x; i < (BATCH * HID) / 2; i += 256) hz[i] = 0u;
  }
  const int tid = blockIdx.x * 256 + threadIdx.x;
  const int nth = gridDim.x * 256;
  for (int i = tid; i < (M1 * IN) / 4; i += nth)    cvtstore4(x,  xb,  i);
  for (int i = tid; i < (G3 * IN) / 4; i += nth)    cvtstore4(w2, w2b, i);
  for (int i = tid; i < (OUTF * HID) / 4; i += nth) cvtstore4(fw, fcb, i);
}

// ---------------- kernel 1: x_proj = x @ x2h_w^T + x2h_b (bf16 out) --------
// Wave task: one 16x64 strip (M-tile x 4 N-tiles), pure bf16 loads.
__global__ __launch_bounds__(256) void k_xproj(
    const unsigned short* __restrict__ xb_u16,
    const unsigned short* __restrict__ w2b_u16,
    const float* __restrict__ bias, unsigned short* __restrict__ xpb_u16) {
  const __bf16* xb  = reinterpret_cast<const __bf16*>(xb_u16);
  const __bf16* w2b = reinterpret_cast<const __bf16*>(w2b_u16);
  __bf16* xpb = reinterpret_cast<__bf16*>(xpb_u16);

  const int lane = threadIdx.x & 31;
  const int wid  = threadIdx.x >> 5;
  const int gid  = blockIdx.x * 8 + wid;          // 0..49151
  const int mt   = gid / 24;
  const int ns   = gid - mt * 24;
  const int m0   = mt * 16;
  const int g0   = ns * 64;
  const int half = (lane < 16) ? 0 : 1;
  const int lrow = lane & 15;

  v8f a0 = {}, a1 = {}, a2 = {}, a3 = {};
  const __bf16* xrow = xb + (size_t)(m0 + lrow) * IN;
  const __bf16* wr0 = w2b + (size_t)(g0 +  0 + lrow) * IN;
  const __bf16* wr1 = w2b + (size_t)(g0 + 16 + lrow) * IN;
  const __bf16* wr2 = w2b + (size_t)(g0 + 32 + lrow) * IN;
  const __bf16* wr3 = w2b + (size_t)(g0 + 48 + lrow) * IN;

#pragma unroll
  for (int kk = 0; kk < IN / 32; ++kk) {
    const int k0 = kk * 32;
    v16bf a;
    ld16(xrow + k0 + half * 16, a);
    const int kb = k0 + half * 16;
    v16bf b0, b1, b2, b3;
    ld16(wr0 + kb, b0);
    ld16(wr1 + kb, b1);
    ld16(wr2 + kb, b2);
    ld16(wr3 + kb, b3);
    a0 = wmma_bf16(a, b0, a0);
    a1 = wmma_bf16(a, b1, a1);
    a2 = wmma_bf16(a, b2, a2);
    a3 = wmma_bf16(a, b3, a3);
  }
  const int N  = lrow;
  const int Mb = half * 8;
  const float bv0 = bias[g0 +  0 + N];
  const float bv1 = bias[g0 + 16 + N];
  const float bv2 = bias[g0 + 32 + N];
  const float bv3 = bias[g0 + 48 + N];
#pragma unroll
  for (int r = 0; r < 8; ++r) {
    __bf16* orow = xpb + (size_t)(m0 + Mb + r) * G3 + g0;
    orow[ 0 + N] = (__bf16)(a0[r] + bv0);
    orow[16 + N] = (__bf16)(a1[r] + bv1);
    orow[32 + N] = (__bf16)(a2[r] + bv2);
    orow[48 + N] = (__bf16)(a3[r] + bv3);
  }
}

// ---------------- kernel 2: persistent GRU recurrence ----------------
// 16 WGs x 256 threads. Each wave owns one (16-row, 16-col) tile for all 3
// gates; its h2h_w slice lives in 384 VGPRs for the whole kernel. A-fragments
// are software-pipelined (double buffer) so loads for step kk+1 issue before
// the WMMAs of step kk. Global sense barrier (atomic counter) per timestep;
// h double-buffered bf16; next step's x_proj lines prefetched.
__global__ __launch_bounds__(256) void k_gru(
    const unsigned short* __restrict__ xpb_u16,
    const float* __restrict__ h2h_w, const float* __restrict__ h2h_b,
    unsigned short* __restrict__ hbuf_u16, float* __restrict__ act,
    unsigned short* __restrict__ actb_u16, unsigned* __restrict__ cnt) {
  const __bf16* xpb = reinterpret_cast<const __bf16*>(xpb_u16);
  __bf16* hbuf = reinterpret_cast<__bf16*>(hbuf_u16);
  __bf16* actb = reinterpret_cast<__bf16*>(actb_u16);

  const int lane = threadIdx.x & 31;
  const int wid  = threadIdx.x >> 5;               // 0..7
  const int m0   = (wid & 3) * 16;                 // batch-row tile
  const int c0   = blockIdx.x * 32 + (wid >> 2) * 16;  // gate-relative cols
  const int half = (lane < 16) ? 0 : 1;
  const int lrow = lane & 15;
  const int N    = lrow;
  const int Mb   = half * 8;
  const int cg   = c0 + N;                         // this lane's h column

  // Preload B fragments (h2h_w rows g*512 + c0 + lrow) into registers.
  v16bf Bf[3][16];
#pragma unroll
  for (int g = 0; g < 3; ++g) {
    const float* wr = h2h_w + (size_t)(g * HID + c0 + lrow) * HID;
#pragma unroll
    for (int kk = 0; kk < 16; ++kk)
      cvt16(wr + kk * 32 + half * 16, Bf[g][kk]);
  }
  const float hb_r = h2h_b[cg];
  const float hb_u = h2h_b[HID + cg];
  const float hb_n = h2h_b[2 * HID + cg];

  float hreg[8] = {0.f, 0.f, 0.f, 0.f, 0.f, 0.f, 0.f, 0.f};

  for (int t = 0; t < T_STEPS; ++t) {
    const __bf16* hcur  = hbuf + (size_t)(t & 1) * (BATCH * HID);
    __bf16*       hnext = hbuf + (size_t)((t + 1) & 1) * (BATCH * HID);

    // Warm next step's x_proj lines into cache while the GEMM runs.
    if (t + 1 < T_STEPS) {
      const __bf16* xpn = xpb + (size_t)(t + 1) * BATCH * G3 + cg;
#pragma unroll
      for (int r = 0; r < 8; ++r) {
        const __bf16* p = xpn + (size_t)(m0 + Mb + r) * G3;
        __builtin_prefetch(p, 0, 3);
        __builtin_prefetch(p + HID, 0, 3);
        __builtin_prefetch(p + 2 * HID, 0, 3);
      }
    }

    v8f aR = {}, aU = {}, aN = {};
    const __bf16* hrow = hcur + (size_t)(m0 + lrow) * HID;
    const int kofs = half * 16;

    // Software-pipelined K loop: issue loads for kk+1 before WMMAs of kk.
    v16bf a_cur, a_nxt;
    ld8(hrow + kofs,      a_cur, 0);
    ld8(hrow + kofs + 8,  a_cur, 8);
#pragma unroll
    for (int kk = 0; kk < 16; ++kk) {
      if (kk + 1 < 16) {
        const int kb = (kk + 1) * 32 + kofs;
        ld8(hrow + kb,      a_nxt, 0);
        ld8(hrow + kb + 8,  a_nxt, 8);
      }
      aR = wmma_bf16(a_cur, Bf[0][kk], aR);
      aU = wmma_bf16(a_cur, Bf[1][kk], aU);
      aN = wmma_bf16(a_cur, Bf[2][kk], aN);
      a_cur = a_nxt;
    }

#pragma unroll
    for (int r = 0; r < 8; ++r) {
      const int b = m0 + Mb + r;
      const __bf16* xpe = xpb + ((size_t)t * BATCH + b) * G3 + cg;
      const float x_r = (float)xpe[0];
      const float x_u = (float)xpe[HID];
      const float x_n = (float)xpe[2 * HID];
      const float h_r = aR[r] + hb_r;
      const float h_u = aU[r] + hb_u;
      const float h_n = aN[r] + hb_n;
      const float reset = sigmoidf_(x_r + h_r);
      const float upd   = sigmoidf_(x_u + h_u);
      const float nv    = tanhf_(x_n + reset * h_n);
      const float hy    = upd * hreg[r] + (1.0f - upd) * nv;
      hreg[r] = hy;
      hnext[(size_t)b * HID + cg] = (__bf16)hy;
      act [((size_t)t * BATCH + b) * HID + cg] = hy;
      actb[((size_t)t * BATCH + b) * HID + cg] = (__bf16)hy;
    }

    __threadfence();
    __syncthreads();
    if (threadIdx.x == 0) {
      atomicAdd(cnt, 1u);
      const unsigned target = (unsigned)(NWG2 * (t + 1));
      while (atomicAdd(cnt, 0u) < target) { __builtin_amdgcn_s_sleep(4); }
    }
    __syncthreads();
    __threadfence();
  }
}

// ---------------- kernel 3: out = act @ fc_w^T + fc_b ----------------
// Wave task: one 16x64 strip (full O dimension), pure bf16 loads.
__global__ __launch_bounds__(256) void k_fc(
    const unsigned short* __restrict__ actb_u16,
    const unsigned short* __restrict__ fcb_u16,
    const float* __restrict__ fb, float* __restrict__ out) {
  const __bf16* actb = reinterpret_cast<const __bf16*>(actb_u16);
  const __bf16* fcb  = reinterpret_cast<const __bf16*>(fcb_u16);

  const int lane = threadIdx.x & 31;
  const int wid  = threadIdx.x >> 5;
  const int gid  = blockIdx.x * 8 + wid;   // 0..2047 m-tile
  const int m0   = gid * 16;
  const int half = (lane < 16) ? 0 : 1;
  const int lrow = lane & 15;

  v8f a0 = {}, a1 = {}, a2 = {}, a3 = {};
  const __bf16* arow = actb + (size_t)(m0 + lrow) * HID;
  const __bf16* wr0 = fcb + (size_t)( 0 + lrow) * HID;
  const __bf16* wr1 = fcb + (size_t)(16 + lrow) * HID;
  const __bf16* wr2 = fcb + (size_t)(32 + lrow) * HID;
  const __bf16* wr3 = fcb + (size_t)(48 + lrow) * HID;

#pragma unroll
  for (int kk = 0; kk < HID / 32; ++kk) {
    const int k0 = kk * 32;
    v16bf a;
    ld16(arow + k0 + half * 16, a);
    const int kb = k0 + half * 16;
    v16bf b0, b1, b2, b3;
    ld16(wr0 + kb, b0);
    ld16(wr1 + kb, b1);
    ld16(wr2 + kb, b2);
    ld16(wr3 + kb, b3);
    a0 = wmma_bf16(a, b0, a0);
    a1 = wmma_bf16(a, b1, a1);
    a2 = wmma_bf16(a, b2, a2);
    a3 = wmma_bf16(a, b3, a3);
  }
  const int N  = lrow;
  const int Mb = half * 8;
  const float bv0 = fb[ 0 + N];
  const float bv1 = fb[16 + N];
  const float bv2 = fb[32 + N];
  const float bv3 = fb[48 + N];
#pragma unroll
  for (int r = 0; r < 8; ++r) {
    float* orow = out + (size_t)(m0 + Mb + r) * OUTF;
    orow[ 0 + N] = a0[r] + bv0;
    orow[16 + N] = a1[r] + bv1;
    orow[32 + N] = a2[r] + bv2;
    orow[48 + N] = a3[r] + bv3;
  }
}

// ---------------- launch ----------------
extern "C" void kernel_launch(void* const* d_in, const int* in_sizes, int n_in,
                              void* d_out, int out_size, void* d_ws, size_t ws_size,
                              hipStream_t stream) {
  (void)in_sizes; (void)n_in; (void)out_size; (void)ws_size;
  const float* x     = (const float*)d_in[0];
  const float* x2h_w = (const float*)d_in[1];
  const float* x2h_b = (const float*)d_in[2];
  const float* h2h_w = (const float*)d_in[3];
  const float* h2h_b = (const float*)d_in[4];
  const float* fc_w  = (const float*)d_in[5];
  const float* fc_b  = (const float*)d_in[6];

  float* out = (float*)d_out;
  float* act = out + OUT_ELEMS;        // rnn_activity part of d_out

  char* ws = (char*)d_ws;
  unsigned*       cnt  = (unsigned*)ws;
  unsigned short* hbuf = (unsigned short*)(ws + H_OFF);
  unsigned short* xb   = (unsigned short*)(ws + XB_OFF);
  unsigned short* w2b  = (unsigned short*)(ws + W2B_OFF);
  unsigned short* fcb  = (unsigned short*)(ws + FCB_OFF);
  unsigned short* actb = (unsigned short*)(ws + ACTB_OFF);
  unsigned short* xpb  = (unsigned short*)(ws + XPB_OFF);

  // 0) one-shot bf16 conversion of x / x2h_w / fc_w, zero-init h & counter
  k_prep<<<1024, 256, 0, stream>>>(x, x2h_w, fc_w, xb, w2b, fcb,
                                   cnt, (unsigned*)hbuf);
  // 1) input projection for all timesteps (bf16 in, bf16 out)
  k_xproj<<<(M1 / 16) * 24 / 8, 256, 0, stream>>>(xb, w2b, x2h_b, xpb);
  // 2) persistent recurrent kernel: 16 WGs, global barrier per step
  k_gru<<<NWG2, 256, 0, stream>>>(xpb, h2h_w, h2h_b, hbuf, act, actb, cnt);
  // 3) output projection (bf16 in, f32 out)
  k_fc<<<(M1 / 16) / 8, 256, 0, stream>>>(actb, fcb, fc_b, out);
}
